// QuantumDiT_9380208575228
// MI455X (gfx1250) — matrix operand order repaired and checked by reading, not compiled, not run on hardware
//
#include <hip/hip_runtime.h>

// ---------------- types / constants ----------------
typedef __attribute__((ext_vector_type(8)))  __bf16 v8bf;
typedef __attribute__((ext_vector_type(16))) __bf16 v16bf;
typedef __attribute__((ext_vector_type(8)))  float  v8f;
typedef __attribute__((ext_vector_type(4)))  int    v4i;

typedef __attribute__((address_space(1))) v4i GA_v4i;   // global int4*
typedef __attribute__((address_space(3))) v4i LA_v4i;   // LDS int4*

#if __has_builtin(__builtin_amdgcn_global_load_async_to_lds_b128)
#define HAVE_ASYNC_LDS 1
#else
#define HAVE_ASYNC_LDS 0
#endif

static __device__ inline void wait_async_le4() {
#if HAVE_ASYNC_LDS
#if __has_builtin(__builtin_amdgcn_s_wait_asynccnt)
    __builtin_amdgcn_s_wait_asynccnt(4);
#else
    asm volatile("s_wait_asynccnt 4" ::: "memory");
#endif
#endif
}
static __device__ inline void wait_async_0() {
#if HAVE_ASYNC_LDS
#if __has_builtin(__builtin_amdgcn_s_wait_asynccnt)
    __builtin_amdgcn_s_wait_asynccnt(0);
#else
    asm volatile("s_wait_asynccnt 0" ::: "memory");
#endif
#endif
}

constexpr int Bc = 16, Sc = 256, Dm = 1024, Hh = 16, Ff = 4096, Lc = 6, Vv = 32;
constexpr int HDc = 64, BSc = Bc * Sc, D3 = 3 * Dm, D4 = 4 * Dm, D6 = 6 * Dm;

union V16U { v16bf v; struct { v8bf lo, hi; } s; };
static __device__ inline v16bf mk16(v8bf lo, v8bf hi) { V16U u; u.s.lo = lo; u.s.hi = hi; return u.v; }

static __device__ inline v8f wmma_bf16(v16bf a, v16bf b, v8f c) {
    // (neg_a, A, neg_b, B, c_mod, C, reuse_a, reuse_b)
    return __builtin_amdgcn_wmma_f32_16x16x32_bf16(false, a, false, b, (short)0, c, false, false);
}

// ---------------- tiled WMMA GEMM: C[M,N] = A[M,K](bf16) x Bt[N,K](bf16, pre-transposed) ----
// Block = 256 threads (8 waves). Tile 128x128x32, double-buffered async-to-LDS pipeline.
// Wave tile 64x32 (4x2 of 16x16).
template <bool OUT_BF16>
__global__ __launch_bounds__(256) void gemm_wmma(
    const __bf16* __restrict__ Ab, const __bf16* __restrict__ Btb, void* __restrict__ Cv,
    int M, int N, int K, long long sA, long long sB, long long sC, int ldc)
{
    (void)M; (void)N;
    const int zb = blockIdx.z;
    const __bf16* A  = Ab  + (size_t)zb * sA;
    const __bf16* Bt = Btb + (size_t)zb * sB;   // [N][K]
    const int tid = threadIdx.x, wid = tid >> 5, lane = tid & 31, lr = lane & 15, hl = lane >> 4;
    const int m_blk = blockIdx.y * 128, n_blk = blockIdx.x * 128;
    const int wm = (wid >> 2) * 64, wn = (wid & 3) * 32;

    __shared__ __bf16 As[2][128][40];   // A tile, row-major, padded row (80B)
    __shared__ __bf16 Bs[2][128][40];   // B tile as Bt rows: [n][k], padded row

    const int rowA = tid >> 2, ccA = (tid & 3) * 8;        // thread's 2 chunks: rowA, rowA+64
    // fill buffer `bi` with K-slice k0 (2 A chunks + 2 B chunks per thread)
    auto fill = [&](int k0, int bi) {
#pragma unroll
        for (int i = 0; i < 2; i++) {
            int row = rowA + i * 64;
            const __bf16* ga = A  + (size_t)(m_blk + row) * K + k0 + ccA;
            const __bf16* gb = Bt + (size_t)(n_blk + row) * K + k0 + ccA;
#if HAVE_ASYNC_LDS
            __builtin_amdgcn_global_load_async_to_lds_b128((GA_v4i*)ga, (LA_v4i*)&As[bi][row][ccA], 0, 0);
            __builtin_amdgcn_global_load_async_to_lds_b128((GA_v4i*)gb, (LA_v4i*)&Bs[bi][row][ccA], 0, 0);
#else
            *(v8bf*)&As[bi][row][ccA] = *(const v8bf*)ga;
            *(v8bf*)&Bs[bi][row][ccA] = *(const v8bf*)gb;
#endif
        }
    };

    v8f acc[4][2] = {};
    fill(0, 0);

    for (int k0 = 0; k0 < K; k0 += 32) {
        const int cur = (k0 >> 5) & 1;
        const bool more = (k0 + 32) < K;
        if (more) fill(k0 + 32, cur ^ 1);   // prefetch next tile (async, stays in flight)
        if (more) wait_async_le4();         // oldest group (current tile) has landed
        else      wait_async_0();
        __syncthreads();

        // A fragments: lane<16 -> K {0..7,16..23}, lane>=16 -> K {8..15,24..31}
        v16bf af[4];
#pragma unroll
        for (int ms = 0; ms < 4; ms++) {
            const __bf16* ap = &As[cur][wm + ms * 16 + lr][hl * 8];
            af[ms] = mk16(*(const v8bf*)ap, *(const v8bf*)(ap + 16));
        }
        // B fragments: lane = column (row of Bt), lane<16 -> K {0..7,16..23}, lane>=16 -> K {8..15,24..31}
#pragma unroll
        for (int ns = 0; ns < 2; ns++) {
            const __bf16* bp = &Bs[cur][wn + ns * 16 + lr][hl * 8];
            v16bf bfr = mk16(*(const v8bf*)bp, *(const v8bf*)(bp + 16));
#pragma unroll
            for (int ms = 0; ms < 4; ms++) acc[ms][ns] = wmma_bf16(af[ms], bfr, acc[ms][ns]);
        }
        __syncthreads();
    }

    // C layout: lane = column, VGPR r -> row r (+8 for lanes 16..31)
#pragma unroll
    for (int ms = 0; ms < 4; ms++)
#pragma unroll
        for (int ns = 0; ns < 2; ns++)
#pragma unroll
            for (int r = 0; r < 8; r++) {
                int row = m_blk + wm + ms * 16 + r + hl * 8;
                int col = n_blk + wn + ns * 16 + lr;
                size_t idx = (size_t)row * ldc + col;
                if constexpr (OUT_BF16) (((__bf16*)Cv) + (size_t)zb * sC)[idx] = (__bf16)acc[ms][ns][r];
                else                    (((float*)Cv)  + (size_t)zb * sC)[idx] = acc[ms][ns][r];
            }
}

// ---------------- fused 4-branch attention, one block per (head,batch,branch) ----------------
// qkv: bf16 [4][B*S][3D], inner col = j*Dm + h*HD + hd.  ctx out: bf16 [4][B*S][Dm]
__global__ __launch_bounds__(256) void attn_wmma(const __bf16* __restrict__ qkv,
                                                 __bf16* __restrict__ ctx)
{
    extern __shared__ __bf16 sm[];
    __bf16* Ks = sm;                 // [256][72]   K tile (row-major)
    __bf16* Vs = Ks + 256 * 72;      // [64][264]   V transposed: Vs[hd][s]
    __bf16* Ps = Vs + 64 * 264;      // [8][16][264] per-wave P tiles
    const int h = blockIdx.x, b = blockIdx.y, z = blockIdx.z;
    const int tid = threadIdx.x, wid = tid >> 5, lane = tid & 31, lr = lane & 15, hl = lane >> 4;
    const size_t rowbase = (size_t)z * BSc + (size_t)b * Sc;

    // stage K (async, contiguous) and V (manual transpose) into LDS
#pragma unroll
    for (int i = 0; i < 8; i++) {
        int c = tid + i * 256;               // 0..2047
        int s = c >> 3, c8 = (c & 7) * 8;
        const __bf16* base = qkv + (rowbase + s) * D3 + h * HDc + c8;
#if HAVE_ASYNC_LDS
        __builtin_amdgcn_global_load_async_to_lds_b128((GA_v4i*)(base + Dm), (LA_v4i*)&Ks[s * 72 + c8], 0, 0);
#else
        *(v8bf*)&Ks[s * 72 + c8] = *(const v8bf*)(base + Dm);
#endif
        v8bf vv = *(const v8bf*)(base + 2 * Dm);                      // V
#pragma unroll
        for (int e = 0; e < 8; e++) Vs[(c8 + e) * 264 + s] = vv[e];
    }
    wait_async_0();
    __syncthreads();

    for (int pass = 0; pass < 2; pass++) {
        const int q0 = pass * 128 + wid * 16;   // 16 q-rows per wave
        v8f acc[16] = {};
        // scores = q . k^T  (K-dim = HD = 64 -> two k-steps)
#pragma unroll
        for (int c = 0; c < 2; c++) {
            const __bf16* qp = qkv + (rowbase + q0 + lr) * D3 + h * HDc + c * 32 + hl * 8;
            v16bf af = mk16(*(const v8bf*)qp, *(const v8bf*)(qp + 16));
#pragma unroll
            for (int t = 0; t < 16; t++) {
                const __bf16* bp = &Ks[(t * 16 + lr) * 72 + c * 32 + hl * 8];
                v16bf bfr = mk16(*(const v8bf*)bp, *(const v8bf*)(bp + 16));
                acc[t] = wmma_bf16(af, bfr, acc[t]);
            }
        }
        // scale + analytic branch bias + softmax (rows live in VGPR r, cols in lanes)
        __bf16* Pw = Ps + wid * 16 * 264;
#pragma unroll
        for (int r = 0; r < 8; r++) {
            int row = q0 + r + hl * 8;
            float mx = -3.0e38f;
#pragma unroll
            for (int t = 0; t < 16; t++) {
                int col = t * 16 + lr;
                int dd = row - col; if (dd < 0) dd = -dd;
                float bias;
                if      (z == 0) bias = -0.05f * (float)dd;
                else if (z == 1) bias = (dd <= 1) ? 0.f : -1e9f;
                else if (z == 2) bias = ((row >> 1) == (col >> 1)) ? 0.f : -1e9f;
                else             bias = 0.f;
                float v = acc[t][r] * 0.125f + bias;
                acc[t][r] = v;
                mx = fmaxf(mx, v);
            }
            mx = fmaxf(mx, __shfl_xor(mx, 1, 16));
            mx = fmaxf(mx, __shfl_xor(mx, 2, 16));
            mx = fmaxf(mx, __shfl_xor(mx, 4, 16));
            mx = fmaxf(mx, __shfl_xor(mx, 8, 16));
            float sum = 0.f;
#pragma unroll
            for (int t = 0; t < 16; t++) { float pv = __expf(acc[t][r] - mx); acc[t][r] = pv; sum += pv; }
            sum += __shfl_xor(sum, 1, 16); sum += __shfl_xor(sum, 2, 16);
            sum += __shfl_xor(sum, 4, 16); sum += __shfl_xor(sum, 8, 16);
            float inv = 1.f / sum;
            int rl = r + hl * 8;
#pragma unroll
            for (int t = 0; t < 16; t++) Pw[rl * 264 + t * 16 + lr] = (__bf16)(acc[t][r] * inv);
        }
        // ctx = P[16x256] . V[256x64]
        v8f cacc[4] = {};
#pragma unroll
        for (int kk = 0; kk < 8; kk++) {
            const __bf16* ap = Pw + lr * 264 + kk * 32 + hl * 8;
            v16bf af = mk16(*(const v8bf*)ap, *(const v8bf*)(ap + 16));
#pragma unroll
            for (int t = 0; t < 4; t++) {
                const __bf16* bp = &Vs[(t * 16 + lr) * 264 + kk * 32 + hl * 8];
                v16bf bfr = mk16(*(const v8bf*)bp, *(const v8bf*)(bp + 16));
                cacc[t] = wmma_bf16(af, bfr, cacc[t]);
            }
        }
#pragma unroll
        for (int t = 0; t < 4; t++)
#pragma unroll
            for (int r = 0; r < 8; r++) {
                int row = q0 + r + hl * 8;
                ctx[(rowbase + row) * Dm + h * HDc + t * 16 + lr] = (__bf16)cacc[t][r];
            }
    }
}

// ---------------- elementwise / small kernels ----------------
// transpose-convert: d[m][n][k] = (bf16) s[m][k][n]   (weights -> column-major bf16)
__global__ void k_cvt_bf16_T(const float* __restrict__ s, __bf16* __restrict__ d,
                             int K, int N, int nmat) {
    size_t total = (size_t)nmat * K * N;
    size_t i = (size_t)blockIdx.x * blockDim.x + threadIdx.x, st = (size_t)gridDim.x * blockDim.x;
    size_t kn = (size_t)K * N;
    for (; i < total; i += st) {
        size_t m = i / kn, r = i % kn;
        size_t n = r / (size_t)K, k = r % (size_t)K;
        d[i] = (__bf16)s[m * kn + k * (size_t)N + n];
    }
}
__global__ void k_copy_f32(const float* __restrict__ s, float* __restrict__ d, size_t n) {
    size_t i = (size_t)blockIdx.x * blockDim.x + threadIdx.x, st = (size_t)gridDim.x * blockDim.x;
    for (; i < n; i += st) d[i] = s[i];
}
__global__ void k_silu(const float* __restrict__ s, float* __restrict__ d, size_t n) {
    size_t i = (size_t)blockIdx.x * blockDim.x + threadIdx.x, st = (size_t)gridDim.x * blockDim.x;
    for (; i < n; i += st) { float v = s[i]; d[i] = v / (1.f + __expf(-v)); }
}
// out[bb,j] = sum_k A[bb,k]*W[k,j] + bias[j]   (tiny M)
__global__ void k_small_gemm_bias(const float* __restrict__ A, const float* __restrict__ W,
                                  const float* __restrict__ bias, float* __restrict__ out, int K, int N) {
    int j = blockIdx.x * blockDim.x + threadIdx.x, bb = blockIdx.y;
    if (j >= N) return;
    float s = bias ? bias[j] : 0.f;
    for (int k = 0; k < K; k++) s += A[(size_t)bb * K + k] * W[(size_t)k * N + j];
    out[(size_t)bb * N + j] = s;
}
// LN + AdaLN modulation -> bf16
__global__ __launch_bounds__(256) void k_ln_mod(const float* __restrict__ x, const float* __restrict__ w,
                                                const float* __restrict__ b, const float* __restrict__ a,
                                                int shc, int scc, __bf16* __restrict__ out) {
    __shared__ float red[256];
    __shared__ float s_mu, s_inv;
    const int t = blockIdx.x, tid = threadIdx.x;
    const float* xr = x + (size_t)t * Dm;
    float s = 0.f;
    for (int d = tid; d < Dm; d += 256) s += xr[d];
    red[tid] = s; __syncthreads();
    for (int o = 128; o > 0; o >>= 1) { if (tid < o) red[tid] += red[tid + o]; __syncthreads(); }
    if (tid == 0) s_mu = red[0] * (1.f / Dm);
    __syncthreads();
    float mu = s_mu, v = 0.f;
    for (int d = tid; d < Dm; d += 256) { float df = xr[d] - mu; v += df * df; }
    red[tid] = v; __syncthreads();
    for (int o = 128; o > 0; o >>= 1) { if (tid < o) red[tid] += red[tid + o]; __syncthreads(); }
    if (tid == 0) s_inv = rsqrtf(red[0] * (1.f / Dm) + 1e-6f);
    __syncthreads();
    float inv = s_inv;
    const int bb = t / Sc;
    const float* ash = a + (size_t)bb * D6 + (size_t)shc * Dm;
    const float* asc = a + (size_t)bb * D6 + (size_t)scc * Dm;
    for (int d = tid; d < Dm; d += 256) {
        float xn = (xr[d] - mu) * inv * w[d] + b[d];
        out[(size_t)t * Dm + d] = (__bf16)(xn * (1.f + asc[d]) + ash[d]);
    }
}
__global__ __launch_bounds__(256) void k_final_ln(const float* __restrict__ x, const float* __restrict__ w,
                                                  const float* __restrict__ b, float* __restrict__ out) {
    __shared__ float red[256];
    __shared__ float s_mu, s_inv;
    const int t = blockIdx.x, tid = threadIdx.x;
    const float* xr = x + (size_t)t * Dm;
    float s = 0.f;
    for (int d = tid; d < Dm; d += 256) s += xr[d];
    red[tid] = s; __syncthreads();
    for (int o = 128; o > 0; o >>= 1) { if (tid < o) red[tid] += red[tid + o]; __syncthreads(); }
    if (tid == 0) s_mu = red[0] * (1.f / Dm);
    __syncthreads();
    float mu = s_mu, v = 0.f;
    for (int d = tid; d < Dm; d += 256) { float df = xr[d] - mu; v += df * df; }
    red[tid] = v; __syncthreads();
    for (int o = 128; o > 0; o >>= 1) { if (tid < o) red[tid] += red[tid + o]; __syncthreads(); }
    if (tid == 0) s_inv = rsqrtf(red[0] * (1.f / Dm) + 1e-6f);
    __syncthreads();
    float inv = s_inv;
    for (int d = tid; d < Dm; d += 256) out[(size_t)t * Dm + d] = (xr[d] - mu) * inv * w[d] + b[d];
}
// x += gate[bb, gc*Dm + d] * (y + bias[d])
__global__ void k_resid_gate(float* __restrict__ x, const float* __restrict__ a, int gc,
                             const float* __restrict__ y, const float* __restrict__ bias, size_t n) {
    size_t i = (size_t)blockIdx.x * blockDim.x + threadIdx.x, st = (size_t)gridDim.x * blockDim.x;
    for (; i < n; i += st) {
        int d = (int)(i % Dm);
        size_t t = i / Dm;
        int bb = (int)(t / Sc);
        float yv = y[i] + (bias ? bias[d] : 0.f);
        x[i] += a[(size_t)bb * D6 + (size_t)gc * Dm + d] * yv;
    }
}
__global__ void k_swiglu(const float* __restrict__ h1, const float* __restrict__ h3,
                         __bf16* __restrict__ g, size_t n) {
    size_t i = (size_t)blockIdx.x * blockDim.x + threadIdx.x, st = (size_t)gridDim.x * blockDim.x;
    for (; i < n; i += st) { float v = h1[i]; g[i] = (__bf16)((v / (1.f + __expf(-v))) * h3[i]); }
}
// out[t,v] = dot(xnf[t,:], Wp[:,v]) + bp[v]   (V=32: one warp per token)
__global__ void k_proj(const float* __restrict__ xnf, const float* __restrict__ Wp,
                       const float* __restrict__ bp, float* __restrict__ out) {
    int id = blockIdx.x * blockDim.x + threadIdx.x;
    int t = id / Vv, v = id % Vv;
    float s = bp[v];
    for (int k = 0; k < Dm; k++) s += xnf[(size_t)t * Dm + k] * Wp[(size_t)k * Vv + v];
    out[id] = s;
}

// ---------------- host orchestration ----------------
static constexpr size_t ATTN_SMEM = (size_t)(256 * 72 + 64 * 264 + 8 * 16 * 264) * sizeof(__bf16);

extern "C" void kernel_launch(void* const* d_in, const int* in_sizes, int n_in,
                              void* d_out, int out_size, void* d_ws, size_t ws_size,
                              hipStream_t stream)
{
    (void)in_sizes; (void)n_in; (void)out_size; (void)ws_size;
    const float* x_in  = (const float*)d_in[0];
    const float* t_emb = (const float*)d_in[1];
    const float* ln1_w = (const float*)d_in[2];
    const float* ln1_b = (const float*)d_in[3];
    const float* ln2_w = (const float*)d_in[4];
    const float* ln2_b = (const float*)d_in[5];
    const float* Wqkv  = (const float*)d_in[6];
    const float* Wo    = (const float*)d_in[7];
    const float* Wf    = (const float*)d_in[8];
    const float* bfb   = (const float*)d_in[9];
    const float* W1    = (const float*)d_in[10];
    const float* W3    = (const float*)d_in[11];
    const float* W2    = (const float*)d_in[12];
    const float* Wa    = (const float*)d_in[13];
    const float* ba    = (const float*)d_in[14];
    const float* fn_w  = (const float*)d_in[15];
    const float* fn_b  = (const float*)d_in[16];
    const float* Wp    = (const float*)d_in[17];
    const float* bp    = (const float*)d_in[18];

    char* pa = (char*)d_ws;
    auto alloc = [&](size_t bytes) -> char* { char* r = pa; pa += (bytes + 255) & ~(size_t)255; return r; };
    float*  x    = (float*)alloc((size_t)BSc * Dm * 4);
    float*  st   = (float*)alloc((size_t)Bc * Dm * 4);
    float*  av   = (float*)alloc((size_t)Bc * D6 * 4);
    __bf16* xn   = (__bf16*)alloc((size_t)BSc * Dm * 2);
    __bf16* qkv  = (__bf16*)alloc((size_t)4 * BSc * D3 * 2);
    __bf16* ctx  = (__bf16*)alloc((size_t)4 * BSc * Dm * 2);
    __bf16* cat  = (__bf16*)alloc((size_t)BSc * D4 * 2);
    float*  y    = (float*)alloc((size_t)BSc * Ff * 4);   // attn_out / h1 / mlp_out / xnf
    float*  h3   = (float*)alloc((size_t)BSc * Ff * 4);
    __bf16* gb   = (__bf16*)alloc((size_t)BSc * Ff * 2);
    // pre-transposed bf16 weights: [.., N, K]
    __bf16* Wqkv_b = (__bf16*)alloc((size_t)Lc * 4 * Dm * D3 * 2);
    __bf16* Wo_b   = (__bf16*)alloc((size_t)Lc * 4 * Dm * Dm * 2);
    __bf16* Wf_b   = (__bf16*)alloc((size_t)Lc * D4 * Dm * 2);
    __bf16* W1_b   = (__bf16*)alloc((size_t)Lc * Dm * Ff * 2);
    __bf16* W3_b   = (__bf16*)alloc((size_t)Lc * Dm * Ff * 2);
    __bf16* W2_b   = (__bf16*)alloc((size_t)Lc * Ff * Dm * 2);

    dim3 blk(256);
    k_cvt_bf16_T<<<4096, blk, 0, stream>>>(Wqkv, Wqkv_b, Dm, D3, Lc * 4);
    k_cvt_bf16_T<<<4096, blk, 0, stream>>>(Wo,   Wo_b,   Dm, Dm, Lc * 4);
    k_cvt_bf16_T<<<4096, blk, 0, stream>>>(Wf,   Wf_b,   D4, Dm, Lc);
    k_cvt_bf16_T<<<4096, blk, 0, stream>>>(W1,   W1_b,   Dm, Ff, Lc);
    k_cvt_bf16_T<<<4096, blk, 0, stream>>>(W3,   W3_b,   Dm, Ff, Lc);
    k_cvt_bf16_T<<<4096, blk, 0, stream>>>(W2,   W2_b,   Ff, Dm, Lc);
    k_copy_f32<<<4096, blk, 0, stream>>>(x_in, x, (size_t)BSc * Dm);
    k_silu<<<64, blk, 0, stream>>>(t_emb, st, (size_t)Bc * Dm);

    for (int l = 0; l < Lc; l++) {
        k_small_gemm_bias<<<dim3(D6 / 256, Bc), blk, 0, stream>>>(
            st, Wa + (size_t)l * Dm * D6, ba + (size_t)l * D6, av, Dm, D6);
        k_ln_mod<<<BSc, blk, 0, stream>>>(x, ln1_w + (size_t)l * Dm, ln1_b + (size_t)l * Dm, av, 0, 1, xn);
        // qkv: 4 branches batched, A shared
        gemm_wmma<true><<<dim3(D3 / 128, BSc / 128, 4), blk, 0, stream>>>(
            xn, Wqkv_b + (size_t)l * 4 * Dm * D3, qkv,
            BSc, D3, Dm, 0LL, (long long)Dm * D3, (long long)BSc * D3, D3);
        attn_wmma<<<dim3(Hh, Bc, 4), blk, ATTN_SMEM, stream>>>(qkv, ctx);
        // Wo per branch -> concat into cat at column offset z*Dm (ldc = 4D)
        gemm_wmma<true><<<dim3(Dm / 128, BSc / 128, 4), blk, 0, stream>>>(
            ctx, Wo_b + (size_t)l * 4 * Dm * Dm, cat,
            BSc, Dm, Dm, (long long)BSc * Dm, (long long)Dm * Dm, (long long)Dm, D4);
        // fusion: cat @ Wf -> y (f32)
        gemm_wmma<false><<<dim3(Dm / 128, BSc / 128, 1), blk, 0, stream>>>(
            cat, Wf_b + (size_t)l * D4 * Dm, y, BSc, Dm, D4, 0LL, 0LL, 0LL, Dm);
        k_resid_gate<<<8192, blk, 0, stream>>>(x, av, 2, y, bfb + (size_t)l * Dm, (size_t)BSc * Dm);
        // MLP
        k_ln_mod<<<BSc, blk, 0, stream>>>(x, ln2_w + (size_t)l * Dm, ln2_b + (size_t)l * Dm, av, 3, 4, xn);
        gemm_wmma<false><<<dim3(Ff / 128, BSc / 128, 1), blk, 0, stream>>>(
            xn, W1_b + (size_t)l * Dm * Ff, y, BSc, Ff, Dm, 0LL, 0LL, 0LL, Ff);
        gemm_wmma<false><<<dim3(Ff / 128, BSc / 128, 1), blk, 0, stream>>>(
            xn, W3_b + (size_t)l * Dm * Ff, h3, BSc, Ff, Dm, 0LL, 0LL, 0LL, Ff);
        k_swiglu<<<8192, blk, 0, stream>>>(y, h3, gb, (size_t)BSc * Ff);
        gemm_wmma<false><<<dim3(Dm / 128, BSc / 128, 1), blk, 0, stream>>>(
            gb, W2_b + (size_t)l * Ff * Dm, y, BSc, Dm, Ff, 0LL, 0LL, 0LL, Dm);
        k_resid_gate<<<8192, blk, 0, stream>>>(x, av, 5, y, nullptr, (size_t)BSc * Dm);
    }
    k_final_ln<<<BSc, blk, 0, stream>>>(x, fn_w, fn_b, y);
    k_proj<<<(BSc * Vv) / 256, blk, 0, stream>>>(y, Wp, bp, (float*)d_out);
}
